// HGTDetector_85358180041617
// MI455X (gfx1250) — compile-verified, matrix operand order checked by reading, not run on hardware
//
#include <hip/hip_runtime.h>
#include <hip/hip_bf16.h>
#include <math.h>

#define NUSER 100000
#define NTWEET 200000
#define EMB 128
#define NH 8
#define HD 16
#define E_FF 800000
#define E_PO 200000

typedef __attribute__((ext_vector_type(16))) __bf16 v16bf;
typedef __attribute__((ext_vector_type(8)))  float  v8f;

union Frag32 { uint4 u[2]; v16bf v; };

// ---------------- device helpers ----------------
__device__ __forceinline__ unsigned short f2bf(float f) {
  union { float f; unsigned u; } c; c.f = f;
  unsigned u = c.u + 0x7FFFu + ((c.u >> 16) & 1u);   // round-to-nearest-even
  return (unsigned short)(u >> 16);
}
__device__ __forceinline__ unsigned fkey(float f) {   // monotonic float->uint
  union { float f; unsigned u; } c; c.f = f;
  return c.u ^ ((c.u >> 31) ? 0xFFFFFFFFu : 0x80000000u);
}
__device__ __forceinline__ float fdec(unsigned k) {
  union { unsigned u; float f; } c;
  c.u = k ^ ((k >> 31) ? 0x80000000u : 0xFFFFFFFFu);
  return c.f;
}

// ---------------- WMMA GEMM: C[N x F] = act(A[N x K](bf16) * Wt[F x K]^T(bf16) + bias) ----------------
// One 16x16 C tile per wave; 8 waves / block. K multiple of 32, N,F multiples of 16.
__global__ void gemm_bf16_wmma(const unsigned short* __restrict__ A,
                               const unsigned short* __restrict__ Wt,
                               const float* __restrict__ bias,
                               float* __restrict__ C,
                               int tilesM, int tilesN, int K, int ldC, int act)
{
  int tile = blockIdx.x * 8 + (threadIdx.x >> 5);
  if (tile >= tilesM * tilesN) return;                 // wave-uniform: EXEC all-1 at wmma
  int tm = tile / tilesN, tn = tile % tilesN;
  int lane = threadIdx.x & 31;
  int grp = lane >> 4, l16 = lane & 15;

  const unsigned short* Ar = A  + (size_t)(tm * 16 + l16) * K;
  const unsigned short* Wr = Wt + (size_t)(tn * 16 + l16) * K;

  v8f acc = {};
  for (int k = 0; k < K; k += 32) {
    int k0 = k + grp * 8;                              // per-lane K base (ISA 16-bit A 16x32 layout)
    Frag32 a, b;
    a.u[0] = *(const uint4*)(Ar + k0);
    a.u[1] = *(const uint4*)(Ar + k0 + 16);
    b.u[0] = *(const uint4*)(Wr + k0);
    b.u[1] = *(const uint4*)(Wr + k0 + 16);
    acc = __builtin_amdgcn_wmma_f32_16x16x32_bf16(false, a.v, false, b.v,
                                                  (short)0, acc, false, false);
  }
  float bv = bias ? bias[tn * 16 + l16] : 0.0f;
#pragma unroll
  for (int v = 0; v < 8; ++v) {
    float x = acc[v] + bv;
    if (act == 1) x = x > 0.0f ? x : 0.01f * x;        // leaky_relu
    C[(size_t)(tm * 16 + grp * 8 + v) * ldC + tn * 16 + l16] = x;
  }
}

// ---------------- conversion / staging kernels ----------------
__global__ void f32_to_bf16_k(const float* __restrict__ in, unsigned short* __restrict__ out, int n) {
  int i = blockIdx.x * blockDim.x + threadIdx.x;
  if (i < n) out[i] = f2bf(in[i]);
}
__global__ void strided_convert_k(const float* __restrict__ in, int ld, int off,
                                  unsigned short* __restrict__ out, int n, int cols) {
  int i = blockIdx.x * blockDim.x + threadIdx.x;
  if (i >= n) return;
  int r = i / cols, c = i - r * cols;
  out[i] = f2bf(in[(size_t)r * ld + off + c]);
}
__global__ void transpose_convert_k(const float* __restrict__ W, unsigned short* __restrict__ Wt,
                                    int K, int F) {                  // W[KxF] -> Wt[FxK] bf16
  int i = blockIdx.x * blockDim.x + threadIdx.x;
  if (i >= K * F) return;
  int f = i / K, k = i - f * K;
  Wt[i] = f2bf(W[(size_t)k * F + f]);
}
__global__ void build_blockdiag_k(const float* __restrict__ a /* [H][D][D] */,
                                  unsigned short* __restrict__ Wt /* [128][128] (F x K) */) {
  int i = blockIdx.x * blockDim.x + threadIdx.x;
  if (i >= EMB * EMB) return;
  int r = i / EMB, c = i - r * EMB;                   // r = h*16+f (out col), c = h*16+d (in)
  int h = r / HD, f = r - h * HD;
  unsigned short v = 0;
  if (c / HD == h) { int d = c - h * HD; v = f2bf(a[h * HD * HD + d * HD + f]); }
  Wt[i] = v;
}
__global__ void gelu_convert_k(const float* __restrict__ in, unsigned short* __restrict__ out, int n) {
  int i = blockIdx.x * blockDim.x + threadIdx.x;
  if (i >= n) return;
  float x = in[i];
  float t = tanhf(0.7978845608f * (x + 0.044715f * x * x * x));
  out[i] = f2bf(0.5f * x * (1.0f + t));
}
__global__ void fill_u32_k(unsigned* __restrict__ p, unsigned v, int n) {
  int i = blockIdx.x * blockDim.x + threadIdx.x;
  if (i < n) p[i] = v;
}

// ---------------- encoder (cat/num small GEMMs) ----------------
__global__ void encode_cat_num_k(const float* __restrict__ ux,
                                 const float* __restrict__ cat_w, const float* __restrict__ cat_b,
                                 const float* __restrict__ num_w, const float* __restrict__ num_b,
                                 float* __restrict__ prop) {
  int i = blockIdx.x * blockDim.x + threadIdx.x;
  if (i >= NUSER * 64) return;
  int n = i >> 6, j = i & 63;
  const float* row = ux + (size_t)n * 777;
  float acc;
  if (j < 32) {
    acc = cat_b[j];
#pragma unroll
    for (int t = 0; t < 4; ++t) acc += row[t] * cat_w[t * 32 + j];
  } else {
    int jj = j - 32;
    acc = num_b[jj];
#pragma unroll
    for (int t = 0; t < 5; ++t) acc += row[4 + t] * num_w[t * 32 + jj];
  }
  acc = acc > 0.0f ? acc : 0.01f * acc;
  prop[(size_t)n * EMB + j] = acc;
}

// ---------------- edge-phase kernels (segment softmax attention) ----------------
__global__ void edge_score_max_k(const int* __restrict__ src, const int* __restrict__ dst,
                                 const float* __restrict__ krel, const float* __restrict__ q,
                                 const float* __restrict__ pr, float* __restrict__ score,
                                 unsigned* __restrict__ mkey, int E) {
  int i = blockIdx.x * blockDim.x + threadIdx.x;
  if (i >= E * NH) return;
  int e = i >> 3, h = i & 7;
  const float* kv = krel + (size_t)src[e] * EMB + h * HD;
  const float* qv = q    + (size_t)dst[e] * EMB + h * HD;
  float s = 0.0f;
#pragma unroll
  for (int d = 0; d < HD; ++d) s += kv[d] * qv[d];
  s *= pr[h] * 0.25f;                                  // 1/sqrt(16)
  score[i] = s;
  atomicMax(mkey + (size_t)dst[e] * NH + h, fkey(s));
}
__global__ void edge_expsum_k(const int* __restrict__ dst, const float* __restrict__ score,
                              const unsigned* __restrict__ mkey, float* __restrict__ ebuf,
                              float* __restrict__ sbuf, int E) {
  int i = blockIdx.x * blockDim.x + threadIdx.x;
  if (i >= E * NH) return;
  int e = i >> 3, h = i & 7;
  float m = fdec(mkey[(size_t)dst[e] * NH + h]);
  float ex = __expf(score[i] - m);
  ebuf[i] = ex;
  atomicAdd(sbuf + (size_t)dst[e] * NH + h, ex);
}
__global__ void edge_scatter_k(const int* __restrict__ src, const int* __restrict__ dst,
                               const float* __restrict__ vrel, const float* __restrict__ ebuf,
                               const float* __restrict__ sbuf, float* __restrict__ agg, int E) {
  int i = blockIdx.x * blockDim.x + threadIdx.x;
  if (i >= E * EMB) return;
  int d = i & 15, h = (i >> 4) & 7, e = i >> 7;
  int de = dst[e];
  float alpha = ebuf[e * NH + h] / sbuf[(size_t)de * NH + h];
  atomicAdd(agg + (size_t)de * EMB + h * HD + d,
            vrel[(size_t)src[e] * EMB + h * HD + d] * alpha);
}

// ---------------- skip blend + classifier ----------------
__global__ void blend_k(const float* __restrict__ a, const float* __restrict__ x,
                        const float* __restrict__ skip, float* __restrict__ out, int n) {
  int i = blockIdx.x * blockDim.x + threadIdx.x;
  if (i >= n) return;
  float beta = 1.0f / (1.0f + __expf(-skip[0]));
  out[i] = beta * a[i] + (1.0f - beta) * x[i];
}
__global__ void classify_k(const float* __restrict__ h, const float* __restrict__ w2,
                           const float* __restrict__ b2, float* __restrict__ out, int N) {
  int n = blockIdx.x * blockDim.x + threadIdx.x;
  if (n >= N) return;
  const float* row = h + (size_t)n * EMB;
  float l0 = b2[0], l1 = b2[1];
  for (int j = 0; j < EMB; ++j) { float v = row[j]; l0 += v * w2[j * 2]; l1 += v * w2[j * 2 + 1]; }
  float m = l0 > l1 ? l0 : l1;
  float e0 = __expf(l0 - m), e1 = __expf(l1 - m);
  float inv = 1.0f / (e0 + e1);
  out[(size_t)n * 2]     = e0 * inv;
  out[(size_t)n * 2 + 1] = e1 * inv;
}

static inline int divup(int a, int b) { return (a + b - 1) / b; }

// ---------------- host orchestration ----------------
extern "C" void kernel_launch(void* const* d_in, const int* in_sizes, int n_in,
                              void* d_out, int out_size, void* d_ws, size_t ws_size,
                              hipStream_t stream) {
  (void)in_sizes; (void)n_in; (void)out_size; (void)ws_size;
  const float* user_x  = (const float*)d_in[0];
  const float* tweet_x = (const float*)d_in[1];
  const int* e_follow  = (const int*)d_in[2];
  const int* e_friend  = (const int*)d_in[3];
  const int* e_post    = (const int*)d_in[4];
  const int* e_rev     = (const int*)d_in[5];
  // enc: 6..13, hgt1: 14..43, hgt2: 44..73, cls: 74..77

  char* base = (char*)d_ws; size_t off = 0;
  auto alloc = [&](size_t bytes) -> void* {
    off = (off + 255) & ~(size_t)255; void* p = base + off; off += bytes; return p;
  };

  float* user_cur  = (float*)alloc((size_t)NUSER  * EMB * 4);
  float* tweet_cur = (float*)alloc((size_t)NTWEET * EMB * 4);
  float* user_nxt  = (float*)alloc((size_t)NUSER  * EMB * 4);
  float* tweet_nxt = (float*)alloc((size_t)NTWEET * EMB * 4);
  float* prop      = (float*)alloc((size_t)NUSER  * EMB * 4);
  float* tmpF      = (float*)alloc((size_t)NTWEET * EMB * 4);
  float* qu        = (float*)alloc((size_t)NUSER  * EMB * 4);
  float* qt        = (float*)alloc((size_t)NTWEET * EMB * 4);
  float* krel      = (float*)alloc((size_t)NTWEET * EMB * 4);
  float* vrel      = (float*)alloc((size_t)NTWEET * EMB * 4);
  float* agg_u     = (float*)alloc((size_t)NUSER  * EMB * 4);
  float* agg_t     = (float*)alloc((size_t)NTWEET * EMB * 4);
  unsigned short* xu_bf = (unsigned short*)alloc((size_t)NUSER  * EMB * 2);
  unsigned short* xt_bf = (unsigned short*)alloc((size_t)NTWEET * EMB * 2);
  unsigned short* kbf_u = (unsigned short*)alloc((size_t)NUSER  * EMB * 2);
  unsigned short* kbf_t = (unsigned short*)alloc((size_t)NTWEET * EMB * 2);
  unsigned short* vbf_u = (unsigned short*)alloc((size_t)NUSER  * EMB * 2);
  unsigned short* vbf_t = (unsigned short*)alloc((size_t)NTWEET * EMB * 2);
  unsigned short* abf   = (unsigned short*)alloc((size_t)NUSER  * 768 * 2);  // largest staging
  unsigned short* wbf   = (unsigned short*)alloc((size_t)768 * 64 * 2);      // largest Wt
  float*    score = (float*)alloc((size_t)E_FF * NH * 4);
  float*    ebuf  = (float*)alloc((size_t)E_FF * NH * 4);
  unsigned* mkey  = (unsigned*)alloc((size_t)NTWEET * NH * 4);
  float*    sbuf  = (float*)alloc((size_t)NTWEET * NH * 4);

  auto conv = [&](const float* in, unsigned short* out, int n) {
    f32_to_bf16_k<<<divup(n, 256), 256, 0, stream>>>(in, out, n);
  };
  auto fill = [&](void* p, unsigned v, int n) {
    fill_u32_k<<<divup(n, 256), 256, 0, stream>>>((unsigned*)p, v, n);
  };
  auto tconv = [&](const float* W, int K, int F) {
    transpose_convert_k<<<divup(K * F, 256), 256, 0, stream>>>(W, wbf, K, F);
  };
  auto gemm = [&](const unsigned short* A, const float* bias, float* C,
                  int N, int K, int F, int ldC, int act) {
    int tm = N / 16, tn = F / 16;
    gemm_bf16_wmma<<<divup(tm * tn, 8), 256, 0, stream>>>(A, wbf, bias, C, tm, tn, K, ldC, act);
  };

  // ===== encoder =====
  encode_cat_num_k<<<divup(NUSER * 64, 256), 256, 0, stream>>>(
      user_x, (const float*)d_in[6], (const float*)d_in[7],
      (const float*)d_in[8], (const float*)d_in[9], prop);
  strided_convert_k<<<divup(NUSER * 768, 256), 256, 0, stream>>>(user_x, 777, 9, abf, NUSER * 768, 768);
  tconv((const float*)d_in[10], 768, 64);
  gemm(abf, (const float*)d_in[11], prop + 64, NUSER, 768, 64, EMB, 1);
  conv(prop, abf, NUSER * EMB);
  tconv((const float*)d_in[12], EMB, EMB);
  gemm(abf, (const float*)d_in[13], user_cur, NUSER, EMB, EMB, EMB, 1);
  hipMemcpyAsync(tweet_cur, tweet_x, (size_t)NTWEET * EMB * 4, hipMemcpyDeviceToDevice, stream);

  // ===== HGT layer =====
  auto layer = [&](int pb, const float* xu, const float* xt, float* ou, float* ot) {
    conv(xu, xu_bf, NUSER * EMB);
    conv(xt, xt_bf, NTWEET * EMB);
    struct T { int N; const unsigned short* xbf; unsigned short* kbf; unsigned short* vbf; float* q; int po; };
    T tt[2] = { { NUSER, xu_bf, kbf_u, vbf_u, qu, pb },
                { NTWEET, xt_bf, kbf_t, vbf_t, qt, pb + 9 } };
    for (int i = 0; i < 2; ++i) {
      T& t = tt[i];
      tconv((const float*)d_in[t.po + 0], EMB, EMB);                          // K
      gemm(t.xbf, (const float*)d_in[t.po + 1], tmpF, t.N, EMB, EMB, EMB, 0);
      conv(tmpF, t.kbf, t.N * EMB);
      tconv((const float*)d_in[t.po + 2], EMB, EMB);                          // Q
      gemm(t.xbf, (const float*)d_in[t.po + 3], t.q, t.N, EMB, EMB, EMB, 0);
      tconv((const float*)d_in[t.po + 4], EMB, EMB);                          // V
      gemm(t.xbf, (const float*)d_in[t.po + 5], tmpF, t.N, EMB, EMB, EMB, 0);
      conv(tmpF, t.vbf, t.N * EMB);
    }
    fill(agg_u, 0, NUSER * EMB);
    fill(agg_t, 0, NTWEET * EMB);

    struct R { const int* ei; int E; int s; int d; };
    R rr[4] = { { e_follow, E_FF, 0, 0 }, { e_friend, E_FF, 0, 0 },
                { e_post,   E_PO, 0, 1 }, { e_rev,    E_PO, 1, 0 } };
    for (int r = 0; r < 4; ++r) {
      R& rel = rr[r];
      int pa = pb + 18 + 3 * r;
      const unsigned short* kb = rel.s ? kbf_t : kbf_u;
      const unsigned short* vb = rel.s ? vbf_t : vbf_u;
      int Ns = rel.s ? NTWEET : NUSER;
      int Nd = rel.d ? NTWEET : NUSER;
      const float* qd = rel.d ? qt : qu;
      float* aggd = rel.d ? agg_t : agg_u;
      // head-wise 16x16 rel transforms as one block-diagonal 128x128 WMMA GEMM each
      build_blockdiag_k<<<64, 256, 0, stream>>>((const float*)d_in[pa + 0], wbf);
      gemm(kb, nullptr, krel, Ns, EMB, EMB, EMB, 0);
      build_blockdiag_k<<<64, 256, 0, stream>>>((const float*)d_in[pa + 1], wbf);
      gemm(vb, nullptr, vrel, Ns, EMB, EMB, EMB, 0);
      fill(mkey, 0u, Nd * NH);
      fill(sbuf, 0u, Nd * NH);
      const int* src = rel.ei;
      const int* dst = rel.ei + rel.E;
      edge_score_max_k<<<divup(rel.E * NH, 256), 256, 0, stream>>>(
          src, dst, krel, qd, (const float*)d_in[pa + 2], score, mkey, rel.E);
      edge_expsum_k<<<divup(rel.E * NH, 256), 256, 0, stream>>>(dst, score, mkey, ebuf, sbuf, rel.E);
      edge_scatter_k<<<divup(rel.E * EMB, 256), 256, 0, stream>>>(src, dst, vrel, ebuf, sbuf, aggd, rel.E);
    }

    struct O { int N; const float* agg; const float* x; float* o; int po; };
    O oo[2] = { { NUSER, agg_u, xu, ou, pb }, { NTWEET, agg_t, xt, ot, pb + 9 } };
    for (int i = 0; i < 2; ++i) {
      O& o = oo[i];
      gelu_convert_k<<<divup(o.N * EMB, 256), 256, 0, stream>>>(o.agg, abf, o.N * EMB);
      tconv((const float*)d_in[o.po + 6], EMB, EMB);                          // A
      gemm(abf, (const float*)d_in[o.po + 7], tmpF, o.N, EMB, EMB, EMB, 0);
      blend_k<<<divup(o.N * EMB, 256), 256, 0, stream>>>(
          tmpF, o.x, (const float*)d_in[o.po + 8], o.o, o.N * EMB);
    }
  };

  layer(14, user_cur, tweet_cur, user_nxt, tweet_nxt);
  layer(44, user_nxt, tweet_nxt, user_cur, tweet_cur);

  // ===== classifier =====
  conv(user_cur, abf, NUSER * EMB);
  tconv((const float*)d_in[74], EMB, EMB);
  gemm(abf, (const float*)d_in[75], tmpF, NUSER, EMB, EMB, EMB, 1);
  classify_k<<<divup(NUSER, 256), 256, 0, stream>>>(
      tmpF, (const float*)d_in[76], (const float*)d_in[77], (float*)d_out, NUSER);
}
